// RowSelfAttention_81209241632883
// MI455X (gfx1250) — compile-verified, matrix operand order checked by reading, not run on hardware
//
#include <hip/hip_runtime.h>
#include <hip/hip_bf16.h>

// ---------------------------------------------------------------------------
// RowSelfAttention for MI455X (gfx1250), wave32 + v_wmma_f32_16x16x32_bf16.
// Pipeline: QKV proj -> scores (split-K over rows) -> reduce+softmax ->
//           attn@V -> out proj.  All GEMMs on the bf16 WMMA pipe, f32 accum.
// Double-buffered: next global tile staged in VGPRs while WMMA consumes LDS.
// ---------------------------------------------------------------------------

typedef __bf16 bf16;
typedef __attribute__((ext_vector_type(16))) __bf16 v16bf;
typedef __attribute__((ext_vector_type(8)))  __bf16 v8bf;
typedef __attribute__((ext_vector_type(8)))  float  v8f;

#define EMB 768
#define HEADS 12
#define HDIM 64
#define RR 128
#define CC 256
#define MTOT (RR * CC)          // 32768 flattened token rows
#define LD 40                   // LDS row stride (elems): 80B, 16B-aligned, bank-skewed
#define SPLITK 8                // split-K factor for the scores GEMM

union Frag { v16bf v; v8bf h[2]; };

// 16x32 bf16 WMMA operand fragment from a k-contiguous LDS tile (stride LD).
// Lane layout (ISA 7.12.2): lane L holds row mBase + L%16; half L/16 selects
// k in {hi*8..+7} (elements 0-7) and {16+hi*8..+7} (elements 8-15).
__device__ __forceinline__ Frag load_frag(const bf16* tile, int mBase, int lane) {
    Frag f;
    const bf16* p = tile + (mBase + (lane & 15)) * LD + ((lane >> 4) << 3);
    f.h[0] = *(const v8bf*)p;
    f.h[1] = *(const v8bf*)(p + 16);
    return f;
}

__device__ __forceinline__ v8f wmma_bf16(const Frag& a, const Frag& b, v8f c) {
    return __builtin_amdgcn_wmma_f32_16x16x32_bf16(
        false, a.v, false, b.v, (short)0, c, false, false);
}

__device__ __forceinline__ v8bf cvt8(const float* __restrict__ src) {
    v8bf r;
    #pragma unroll
    for (int j = 0; j < 8; ++j) r[j] = (bf16)src[j];
    return r;
}

// ---------------------------------------------------------------------------
// Kernel 1/5: Out[M x 768] = (A[M x 768] @ W[768 x 768] + bias) * scale
// BM=128, BN=64, KB=32; 8 waves, each a 32x32 tile (2x2 WMMA accumulators).
// Register double-buffering over the K loop.
// ---------------------------------------------------------------------------
template <bool A_F32, bool OUT_F32>
__global__ __launch_bounds__(256)
void gemm768_kernel(const void* __restrict__ Aptr, const float* __restrict__ W,
                    const float* __restrict__ bias, void* __restrict__ Out,
                    float scale) {
    __shared__ bf16 As[128 * LD];   // [m][k]
    __shared__ bf16 Bs[64 * LD];    // [n][k] (W tile transposed)

    const int m0 = blockIdx.x * 128;
    const int n0 = blockIdx.y * 64;
    const int t = threadIdx.x;
    const int wave = t >> 5, lane = t & 31;
    const int wm = (wave & 3) * 32;
    const int wn = (wave >> 2) * 32;

    const int rowA = t >> 1, cbA = (t & 1) * 16;   // A: 2 threads/row, 16 elems each
    const int kB = t >> 3,   nbB = (t & 7) * 8;    // W: 8 threads/k-row, 8 n each

    v8bf a0, a1, w0;   // staged tiles (VGPR double buffer)

    auto stageA = [&](int k0) {
        if (A_F32) {
            const float* src = (const float*)Aptr + (size_t)(m0 + rowA) * EMB + k0 + cbA;
            a0 = cvt8(src); a1 = cvt8(src + 8);
        } else {
            const bf16* src = (const bf16*)Aptr + (size_t)(m0 + rowA) * EMB + k0 + cbA;
            a0 = *(const v8bf*)src; a1 = *(const v8bf*)(src + 8);
        }
    };
    auto stageW = [&](int k0) {
        w0 = cvt8(W + (size_t)(k0 + kB) * EMB + n0 + nbB);
    };

    stageA(0); stageW(0);
    v8f acc[2][2] = {};

    for (int k0 = 0; k0 < EMB; k0 += 32) {
        // commit staged tile to LDS
        *(v8bf*)&As[rowA * LD + cbA]     = a0;
        *(v8bf*)&As[rowA * LD + cbA + 8] = a1;
        #pragma unroll
        for (int j = 0; j < 8; ++j) Bs[(nbB + j) * LD + kB] = w0[j];
        __syncthreads();

        if (k0 + 32 < EMB) { stageA(k0 + 32); stageW(k0 + 32); }  // overlap next loads

        Frag a[2], b[2];
        #pragma unroll
        for (int mi = 0; mi < 2; ++mi) a[mi] = load_frag(As, wm + mi * 16, lane);
        #pragma unroll
        for (int ni = 0; ni < 2; ++ni) b[ni] = load_frag(Bs, wn + ni * 16, lane);
        #pragma unroll
        for (int mi = 0; mi < 2; ++mi)
            #pragma unroll
            for (int ni = 0; ni < 2; ++ni)
                acc[mi][ni] = wmma_bf16(a[mi], b[ni], acc[mi][ni]);
        __syncthreads();
    }

    const int nl = lane & 15, hiv = lane >> 4;
    #pragma unroll
    for (int ni = 0; ni < 2; ++ni) {
        const int gn = n0 + wn + ni * 16 + nl;
        const float bv = bias[gn];
        #pragma unroll
        for (int mi = 0; mi < 2; ++mi)
            #pragma unroll
            for (int v = 0; v < 8; ++v) {
                const int gm = m0 + wm + mi * 16 + v + hiv * 8;
                const float val = (acc[mi][ni][v] + bv) * scale;
                if (OUT_F32) ((float*)Out)[(size_t)gm * EMB + gn] = val;
                else         ((bf16*)Out)[(size_t)gm * EMB + gn] = (bf16)val;
            }
    }
}

// ---------------------------------------------------------------------------
// Kernel 2: split-K scores. S[h][i][j] = sum_{r,d} q[r,i,h,d]*k[r,j,h,d].
// K' = r*64+d = 8192 split SPLITK ways (1024 each) -> 384 workgroups.
// BM=BN=128; 8 waves each own a 32x64 tile (2x4 accumulators).
// Partials go to ws; the softmax kernel reduces them (bit-deterministic).
// ---------------------------------------------------------------------------
__global__ __launch_bounds__(256)
void scores_kernel(const bf16* __restrict__ qb, const bf16* __restrict__ kb,
                   float* __restrict__ part) {
    __shared__ bf16 Qs[128 * LD];
    __shared__ bf16 Ks[128 * LD];

    const int z = blockIdx.z;
    const int h = z / SPLITK;
    const int sk = z % SPLITK;
    const int i0 = blockIdx.x * 128;
    const int j0 = blockIdx.y * 128;
    const int t = threadIdx.x;
    const int wave = t >> 5, lane = t & 31;
    const int wm = (wave & 3) * 32;
    const int wn = (wave >> 2) * 64;

    const int row = t >> 1, cb = (t & 1) * 16;
    v8bf q0, q1, k0v, k1v;

    auto stage = [&](int kk) {
        const int r = kk >> 6;       // row-block r (all 32 k share one r)
        const int d0 = kk & 63;      // 0 or 32 within head dim
        const size_t base = (size_t)h * HDIM + d0 + cb;
        const bf16* qs = qb + (size_t)(r * CC + i0 + row) * EMB + base;
        const bf16* ks = kb + (size_t)(r * CC + j0 + row) * EMB + base;
        q0 = *(const v8bf*)qs;  q1 = *(const v8bf*)(qs + 8);
        k0v = *(const v8bf*)ks; k1v = *(const v8bf*)(ks + 8);
    };

    const int kkBeg = sk * (RR * HDIM / SPLITK);
    const int kkEnd = kkBeg + (RR * HDIM / SPLITK);
    stage(kkBeg);
    v8f acc[2][4] = {};

    for (int kk = kkBeg; kk < kkEnd; kk += 32) {
        *(v8bf*)&Qs[row * LD + cb]     = q0;
        *(v8bf*)&Qs[row * LD + cb + 8] = q1;
        *(v8bf*)&Ks[row * LD + cb]     = k0v;
        *(v8bf*)&Ks[row * LD + cb + 8] = k1v;
        __syncthreads();

        if (kk + 32 < kkEnd) stage(kk + 32);   // overlap next global loads

        Frag a[2], b[4];
        #pragma unroll
        for (int mi = 0; mi < 2; ++mi) a[mi] = load_frag(Qs, wm + mi * 16, lane);
        #pragma unroll
        for (int ni = 0; ni < 4; ++ni) b[ni] = load_frag(Ks, wn + ni * 16, lane);
        #pragma unroll
        for (int mi = 0; mi < 2; ++mi)
            #pragma unroll
            for (int ni = 0; ni < 4; ++ni)
                acc[mi][ni] = wmma_bf16(a[mi], b[ni], acc[mi][ni]);
        __syncthreads();
    }

    const int nl = lane & 15, hiv = lane >> 4;
    #pragma unroll
    for (int mi = 0; mi < 2; ++mi)
        #pragma unroll
        for (int ni = 0; ni < 4; ++ni)
            #pragma unroll
            for (int v = 0; v < 8; ++v) {
                const int gi = i0 + wm + mi * 16 + v + hiv * 8;
                const int gj = j0 + wn + ni * 16 + nl;
                part[((size_t)(sk * HEADS + h) * CC + gi) * CC + gj] = acc[mi][ni][v];
            }
}

// ---------------------------------------------------------------------------
// Kernel 3: reduce SPLITK partials + softmax over j, one block per (h,i) row.
// Writes fp32 probs into d_out's attention slot and a bf16 copy for attn@V.
// ---------------------------------------------------------------------------
__global__ __launch_bounds__(256)
void softmax_kernel(const float* __restrict__ part, float* __restrict__ attn,
                    bf16* __restrict__ probsB) {
    __shared__ float red[256];
    const size_t row = blockIdx.x;           // h*CC + i
    const int t = threadIdx.x;

    float x = 0.0f;
    #pragma unroll
    for (int sk = 0; sk < SPLITK; ++sk)
        x += part[((size_t)sk * HEADS * CC + row) * CC + t];

    red[t] = x; __syncthreads();
    for (int s = 128; s > 0; s >>= 1) {
        if (t < s) red[t] = fmaxf(red[t], red[t + s]);
        __syncthreads();
    }
    const float mx = red[0]; __syncthreads();

    const float e = __expf(x - mx);
    red[t] = e; __syncthreads();
    for (int s = 128; s > 0; s >>= 1) {
        if (t < s) red[t] += red[t + s];
        __syncthreads();
    }
    const float p = e * (1.0f / red[0]);

    attn[row * CC + t] = p;
    probsB[row * CC + t] = (bf16)p;
}

// ---------------------------------------------------------------------------
// Kernel 4: O[r,i,h,d] = sum_j P[h][i][j] * v[r,j,h,d].
// Per (h,r): [256 x 64] GEMM, K=256. BM=128, BN=64 (full head dim).
// V tile transposed into LDS [d][j] so B-fragments read k-contiguously.
// ---------------------------------------------------------------------------
__global__ __launch_bounds__(256)
void attnv_kernel(const bf16* __restrict__ probsB, const bf16* __restrict__ vb,
                  bf16* __restrict__ ob) {
    __shared__ bf16 Ps[128 * LD];
    __shared__ bf16 Vs[64 * LD];    // [d][j]

    const int h = blockIdx.z;
    const int r = blockIdx.y;
    const int i0 = blockIdx.x * 128;
    const int t = threadIdx.x;
    const int wave = t >> 5, lane = t & 31;
    const int wm = (wave & 3) * 32;
    const int wn = (wave >> 2) * 32;

    const int rowP = t >> 1, cbP = (t & 1) * 16;
    const int jV = t >> 3,   dbV = (t & 7) * 8;
    v8bf p0, p1, vv;

    auto stage = [&](int j0) {
        const bf16* ps = probsB + ((size_t)h * CC + i0 + rowP) * CC + j0 + cbP;
        p0 = *(const v8bf*)ps; p1 = *(const v8bf*)(ps + 8);
        vv = *(const v8bf*)(vb + (size_t)(r * CC + j0 + jV) * EMB + h * HDIM + dbV);
    };

    stage(0);
    v8f acc[2][2] = {};

    for (int j0 = 0; j0 < CC; j0 += 32) {
        *(v8bf*)&Ps[rowP * LD + cbP]     = p0;
        *(v8bf*)&Ps[rowP * LD + cbP + 8] = p1;
        #pragma unroll
        for (int jj = 0; jj < 8; ++jj) Vs[(dbV + jj) * LD + jV] = vv[jj];
        __syncthreads();

        if (j0 + 32 < CC) stage(j0 + 32);

        Frag a[2], b[2];
        #pragma unroll
        for (int mi = 0; mi < 2; ++mi) a[mi] = load_frag(Ps, wm + mi * 16, lane);
        #pragma unroll
        for (int ni = 0; ni < 2; ++ni) b[ni] = load_frag(Vs, wn + ni * 16, lane);
        #pragma unroll
        for (int mi = 0; mi < 2; ++mi)
            #pragma unroll
            for (int ni = 0; ni < 2; ++ni)
                acc[mi][ni] = wmma_bf16(a[mi], b[ni], acc[mi][ni]);
        __syncthreads();
    }

    const int nl = lane & 15, hiv = lane >> 4;
    #pragma unroll
    for (int mi = 0; mi < 2; ++mi)
        #pragma unroll
        for (int ni = 0; ni < 2; ++ni)
            #pragma unroll
            for (int v = 0; v < 8; ++v) {
                const int gi = i0 + wm + mi * 16 + v + hiv * 8;
                const int gd = wn + ni * 16 + nl;
                ob[(size_t)(r * CC + gi) * EMB + h * HDIM + gd] = (bf16)acc[mi][ni][v];
            }
}

// ---------------------------------------------------------------------------
extern "C" void kernel_launch(void* const* d_in, const int* in_sizes, int n_in,
                              void* d_out, int out_size, void* d_ws, size_t ws_size,
                              hipStream_t stream) {
    const float* x  = (const float*)d_in[0];
    const float* Wq = (const float*)d_in[1];
    const float* bq = (const float*)d_in[2];
    const float* Wk = (const float*)d_in[3];
    const float* bk = (const float*)d_in[4];
    const float* Wv = (const float*)d_in[5];
    const float* bv = (const float*)d_in[6];
    const float* Wo = (const float*)d_in[7];
    const float* bo = (const float*)d_in[8];

    float* out      = (float*)d_out;                     // [128,256,1,768]
    float* attn_out = out + (size_t)MTOT * EMB;          // [12,1,256,256]

    // workspace: bf16 q/k/v/o + bf16 probs + fp32 split-K partials (~220 MB)
    bf16* qb = (bf16*)d_ws;
    bf16* kb = qb + (size_t)MTOT * EMB;
    bf16* vb = kb + (size_t)MTOT * EMB;
    bf16* ob = vb + (size_t)MTOT * EMB;
    bf16* pb = ob + (size_t)MTOT * EMB;                  // [12,256,256] bf16
    float* part = (float*)(pb + (size_t)HEADS * CC * CC); // [SPLITK,12,256,256] f32

    const float scaling = 0.125f / sqrtf((float)RR);     // HDIM^-0.5 / sqrt(R)

    const dim3 blk(256);
    const dim3 gProj(MTOT / 128, EMB / 64);              // 256 x 12

    gemm768_kernel<true, false><<<gProj, blk, 0, stream>>>(x, Wq, bq, qb, scaling);
    gemm768_kernel<true, false><<<gProj, blk, 0, stream>>>(x, Wk, bk, kb, 1.0f);
    gemm768_kernel<true, false><<<gProj, blk, 0, stream>>>(x, Wv, bv, vb, 1.0f);

    scores_kernel<<<dim3(CC / 128, CC / 128, HEADS * SPLITK), blk, 0, stream>>>(qb, kb, part);

    softmax_kernel<<<dim3(HEADS * CC), blk, 0, stream>>>(part, attn_out, pb);

    attnv_kernel<<<dim3(CC / 128, RR, HEADS), blk, 0, stream>>>(pb, vb, ob);

    gemm768_kernel<false, true><<<gProj, blk, 0, stream>>>(ob, Wo, bo, out, 1.0f);
}
